// MultiHeadAttention_23484881174537
// MI455X (gfx1250) — matrix operand order repaired and checked by reading, not compiled, not run on hardware
//
#include <hip/hip_runtime.h>
#include <hip/hip_bf16.h>

// MI455X / gfx1250 fused multi-head attention forward.
// All matmuls on v_wmma_f32_16x16x32_bf16 (wave32). fp32 in/out, bf16 compute.
// GEMMs use double-buffered LDS fed by CDNA5 async copies
// (global_load_async_to_lds_b128 / ASYNCcnt).

typedef __bf16 bf16;
typedef __attribute__((ext_vector_type(16))) __bf16 v16bf;
typedef __attribute__((ext_vector_type(8)))  float  v8f;
typedef __attribute__((ext_vector_type(4)))  __bf16 v4bf;

constexpr int Bn = 4, Sn = 2048, Dn = 2048, Hn = 16, HDn = 128;
constexpr int Mtot = Bn * Sn;           // 8192 rows in all GEMMs

union FragU { v16bf v; uint4 q[2]; };

// A-fragment (16x32 bf16, M x K): lane l%16 = row M, lane/16 = halfsel.
// Packing per ISA 7.12.2: halfs 0..7 = K[h*8 .. h*8+7], halfs 8..15 = K[16+h*8 ..].
__device__ inline v16bf load_frag_a(const bf16* rowp, int h16) {
  FragU f;
  f.q[0] = *(const uint4*)(rowp + h16 * 8);
  f.q[1] = *(const uint4*)(rowp + 16 + h16 * 8);
  return f.v;
}
// B-fragment (32x16 bf16, K x N) loaded from B^T stored row-major (N x K):
// lane l%16 = column N, halfs 0..7 = K[h*8 ..], halfs 8..15 = K[16+h*8 ..].
__device__ inline v16bf load_frag_b(const bf16* rowp, int h16) {
  const bf16* p = rowp + h16 * 8;
  FragU f;
  f.q[0] = *(const uint4*)(p);
  f.q[1] = *(const uint4*)(p + 16);
  return f.v;
}
__device__ inline v8f wmma_bf16f32(v16bf a, v16bf b, v8f c) {
  return __builtin_amdgcn_wmma_f32_16x16x32_bf16(false, a, false, b, (short)0, c,
                                                 false, false);
}
// Row reductions across the 16 lanes that hold one C/D row (xor 1,2,4,8 stays
// inside each 16-lane half of the wave32).
__device__ inline float rowmax16(float v) {
  v = fmaxf(v, __shfl_xor(v, 1));
  v = fmaxf(v, __shfl_xor(v, 2));
  v = fmaxf(v, __shfl_xor(v, 4));
  v = fmaxf(v, __shfl_xor(v, 8));
  return v;
}
__device__ inline float rowsum16(float v) {
  v += __shfl_xor(v, 1);
  v += __shfl_xor(v, 2);
  v += __shfl_xor(v, 4);
  v += __shfl_xor(v, 8);
  return v;
}

// ---------------- fp32 -> bf16 conversion ----------------
__global__ __launch_bounds__(256) void cvt_f32_bf16(const float* __restrict__ in,
                                                    bf16* __restrict__ out, int n) {
  int i = (blockIdx.x * 256 + threadIdx.x) * 4;
  if (i + 3 < n) {
    float4 v = *(const float4*)(in + i);
    v4bf o;
    o[0] = (bf16)v.x; o[1] = (bf16)v.y; o[2] = (bf16)v.z; o[3] = (bf16)v.w;
    *(v4bf*)(out + i) = o;
  }
}

// Weight convert + transpose: in = W (K x N fp32), out = W^T (N x K bf16).
__global__ __launch_bounds__(256) void cvt_transpose(const float* __restrict__ in,
                                                     bf16* __restrict__ out) {
  __shared__ bf16 t[32][33];
  int bx = blockIdx.x % (Dn / 32), by = blockIdx.x / (Dn / 32);
  int lx = threadIdx.x & 31, ly = threadIdx.x >> 5;   // ly = 0..7
#pragma unroll
  for (int i = 0; i < 4; i++) {
    int k = by * 32 + ly + i * 8;
    t[ly + i * 8][lx] = (bf16)in[k * Dn + bx * 32 + lx];
  }
  __syncthreads();
#pragma unroll
  for (int i = 0; i < 4; i++) {
    int n = bx * 32 + ly + i * 8;
    out[n * Dn + by * 32 + lx] = t[lx][ly + i * 8];
  }
}

// CDNA5 async DMA of one 128x32 bf16 slab of A and of B^T into LDS.
// GVS form: saddr = SGPR64 block base, vaddr = 32-bit byte offset,
// vdst = LDS byte address (low 32 bits of the generic pointer).
__device__ inline void async_stage(const bf16* gA, const bf16* gB,
                                   bf16* lA, bf16* lB, int tid, int k0) {
#pragma unroll
  for (int i = 0; i < 2; i++) {
    int c = tid + i * 256;                 // 512 chunks of 8 bf16 per slab
    int r = c >> 2, kk = (c & 3) * 8;
    unsigned loffA = (unsigned)(uintptr_t)(lA + r * 32 + kk);
    unsigned loffB = (unsigned)(uintptr_t)(lB + r * 32 + kk);
    unsigned goff = (unsigned)((r * Dn + k0 + kk) * (int)sizeof(bf16));
    asm volatile("global_load_async_to_lds_b128 %0, %1, %2"
                 :: "v"(loffA), "v"(goff), "s"(gA) : "memory");
    asm volatile("global_load_async_to_lds_b128 %0, %1, %2"
                 :: "v"(loffB), "v"(goff), "s"(gB) : "memory");
  }
}
__device__ inline void wait_async_all() {
  asm volatile("s_wait_asynccnt 0x0" ::: "memory");
}

// ---------------- WMMA GEMM: C(8192x2048) = A(8192x2048) * B^T(2048x2048) ----
// MODE 0: write bf16, head-split  [b,h,s,hd]   (Q and K)
// MODE 1: write bf16, transposed  [b,h,hd,s]   (V)
// MODE 2: write fp32 + bias, row-major [m,n]   (output projection)
template <int MODE>
__global__ __launch_bounds__(256) void gemm_wmma(const bf16* __restrict__ A,
                                                 const bf16* __restrict__ Bt,
                                                 void* __restrict__ outp,
                                                 const float* __restrict__ bias) {
  constexpr int Kd = Dn;
  __shared__ bf16 As[2][128 * 32];
  __shared__ bf16 Bs[2][128 * 32];

  int tid = threadIdx.x;
  int lane = tid & 31, wave = tid >> 5;
  int h16 = lane >> 4, lane16 = lane & 15;
  int wm = wave >> 1, wn = wave & 1;                 // 4 x 2 wave grid
  int numN = Dn / 128;
  int tileM = blockIdx.x / numN, tileN = blockIdx.x % numN;
  const bf16* Ablk = A + (size_t)tileM * 128 * Kd;
  const bf16* Bblk = Bt + (size_t)tileN * 128 * Kd;

  v8f acc[2][4];
#pragma unroll
  for (int mi = 0; mi < 2; mi++)
#pragma unroll
    for (int ni = 0; ni < 4; ni++)
#pragma unroll
      for (int j = 0; j < 8; j++) acc[mi][ni][j] = 0.0f;

  // prologue: DMA tile 0 into buffer 0
  async_stage(Ablk, Bblk, As[0], Bs[0], tid, 0);

  for (int k0 = 0; k0 < Kd; k0 += 32) {
    int buf = (k0 >> 5) & 1;
    wait_async_all();          // our async copies for this tile have landed
    __syncthreads();           // everyone's copies landed; prev reads retired

    if (k0 + 32 < Kd)          // DMA next tile into the other buffer while
      async_stage(Ablk, Bblk, As[buf ^ 1], Bs[buf ^ 1], tid, k0 + 32);

    v16bf af[2], bfr[4];
#pragma unroll
    for (int mi = 0; mi < 2; mi++)
      af[mi] = load_frag_a(As[buf] + (wm * 32 + mi * 16 + lane16) * 32, h16);
#pragma unroll
    for (int ni = 0; ni < 4; ni++)
      bfr[ni] = load_frag_b(Bs[buf] + (wn * 64 + ni * 16 + lane16) * 32, h16);
#pragma unroll
    for (int mi = 0; mi < 2; mi++)
#pragma unroll
      for (int ni = 0; ni < 4; ni++)
        acc[mi][ni] = wmma_bf16f32(af[mi], bfr[ni], acc[mi][ni]);
  }

  // Epilogue. C/D layout: lane holds column n = lane%16; row m = 8*(lane/16)+j.
#pragma unroll
  for (int mi = 0; mi < 2; mi++) {
#pragma unroll
    for (int ni = 0; ni < 4; ni++) {
      int nn = tileN * 128 + wn * 64 + ni * 16 + lane16;
      int mbase = tileM * 128 + wm * 32 + mi * 16 + 8 * h16;
      float bv = 0.0f;
      if constexpr (MODE == 2) bv = bias[nn];
#pragma unroll
      for (int j = 0; j < 8; j++) {
        int m = mbase + j;
        int b = m >> 11, s = m & (Sn - 1);
        int hh = nn >> 7, d = nn & (HDn - 1);
        float val = acc[mi][ni][j];
        if constexpr (MODE == 0) {
          ((bf16*)outp)[(((size_t)(b * Hn + hh) * Sn) + s) * HDn + d] = (bf16)val;
        } else if constexpr (MODE == 1) {
          ((bf16*)outp)[(((size_t)(b * Hn + hh) * HDn) + d) * Sn + s] = (bf16)val;
        } else {
          ((float*)outp)[(size_t)m * Dn + nn] = val + bv;
        }
      }
    }
  }
}

// ---------------- Flash attention: one wave per 16-row q-tile ----------------
__global__ __launch_bounds__(128) void attn_fused(const bf16* __restrict__ Q,
                                                  const bf16* __restrict__ Kd,
                                                  const bf16* __restrict__ Vt,
                                                  bf16* __restrict__ ctx) {
  __shared__ bf16 Pst[4][16 * 32];          // per-wave P staging (16x32 bf16)
  int lane = threadIdx.x & 31, wave = threadIdx.x >> 5;
  int h16 = lane >> 4, lane16 = lane & 15;
  int tile = blockIdx.x * 4 + wave;         // 8192 q-tiles total
  int q0 = (tile & 127) * 16;
  int h = (tile >> 7) & (Hn - 1);
  int b = tile >> 11;
  const bf16* Qbh = Q + ((size_t)(b * Hn + h) * Sn) * HDn;
  const bf16* Kbh = Kd + ((size_t)(b * Hn + h) * Sn) * HDn;
  const bf16* Vbh = Vt + ((size_t)(b * Hn + h) * HDn) * Sn;
  bf16* Pw = Pst[wave];

  // Q fragments for this 16-row tile: 4 chunks of K=32 over hd=128.
  v16bf qf[4];
#pragma unroll
  for (int c = 0; c < 4; c++)
    qf[c] = load_frag_a(Qbh + (size_t)(q0 + lane16) * HDn + c * 32, h16);

  v8f acc[8];
  float mmax[8], msum[8];
#pragma unroll
  for (int dt = 0; dt < 8; dt++)
#pragma unroll
    for (int j = 0; j < 8; j++) acc[dt][j] = 0.0f;
#pragma unroll
  for (int j = 0; j < 8; j++) { mmax[j] = -1e30f; msum[j] = 0.0f; }

  const float scale = 0.08838834764831845f;  // 1/sqrt(128)

  for (int k0 = 0; k0 <= q0 + 15; k0 += 32) {
    if (k0 + 32 <= q0 + 15)                  // global_prefetch_b8 next K block
      __builtin_prefetch((const void*)(Kbh + (size_t)(k0 + 32 + lane16) * HDn),
                         0, 1);
    // scores: two 16x16 tiles (k columns k0..k0+15 and k0+16..k0+31)
    v8f s0, s1;
#pragma unroll
    for (int j = 0; j < 8; j++) { s0[j] = 0.0f; s1[j] = 0.0f; }
#pragma unroll
    for (int c = 0; c < 4; c++) {
      v16bf kf0 = load_frag_b(Kbh + (size_t)(k0 + lane16) * HDn + c * 32, h16);
      v16bf kf1 = load_frag_b(Kbh + (size_t)(k0 + 16 + lane16) * HDn + c * 32, h16);
      s0 = wmma_bf16f32(qf[c], kf0, s0);
      s1 = wmma_bf16f32(qf[c], kf1, s1);
    }
    // online softmax (per row m = 8*h16+j; its 16 N-values sit across lanes)
#pragma unroll
    for (int j = 0; j < 8; j++) {
      int qrow = q0 + 8 * h16 + j;
      float a0 = s0[j] * scale, a1 = s1[j] * scale;
      if (k0 + lane16 > qrow) a0 = -1e30f;          // causal mask (no branch:
      if (k0 + 16 + lane16 > qrow) a1 = -1e30f;     //  EXEC stays all-1s)
      float tmax = rowmax16(fmaxf(a0, a1));
      float nm = fmaxf(mmax[j], tmax);
      float cf = __expf(mmax[j] - nm);
      float p0 = __expf(a0 - nm), p1 = __expf(a1 - nm);
      msum[j] = msum[j] * cf + rowsum16(p0 + p1);
      mmax[j] = nm;
#pragma unroll
      for (int dt = 0; dt < 8; dt++) acc[dt][j] *= cf;
      int m = 8 * h16 + j;
      Pw[m * 32 + lane16] = (bf16)p0;
      Pw[m * 32 + 16 + lane16] = (bf16)p1;
    }
    // D-layout -> A-fragment relayout through LDS (same wave; order via DScnt)
    asm volatile("s_wait_dscnt 0x0" ::: "memory");
    v16bf pf = load_frag_a(Pw + lane16 * 32, h16);
#pragma unroll
    for (int dt = 0; dt < 8; dt++) {
      v16bf vf = load_frag_b(Vbh + (size_t)(dt * 16 + lane16) * Sn + k0, h16);
      acc[dt] = wmma_bf16f32(pf, vf, acc[dt]);
    }
  }

  // normalize and write ctx in [b, s, h*128+d] row-major (matches "bqhd")
#pragma unroll
  for (int dt = 0; dt < 8; dt++) {
#pragma unroll
    for (int j = 0; j < 8; j++) {
      int m = 8 * h16 + j;
      float o = acc[dt][j] / msum[j];
      ctx[((size_t)(b * Sn + q0 + m) * Dn) + h * HDn + dt * 16 + lane16] = (bf16)o;
    }
  }
}

extern "C" void kernel_launch(void* const* d_in, const int* in_sizes, int n_in,
                              void* d_out, int out_size, void* d_ws, size_t ws_size,
                              hipStream_t stream) {
  const float* x  = (const float*)d_in[0];
  const float* Wq = (const float*)d_in[1];
  const float* Wk = (const float*)d_in[2];
  const float* Wv = (const float*)d_in[3];
  const float* Wo = (const float*)d_in[4];
  const float* bo = (const float*)d_in[5];
  float* out = (float*)d_out;

  // Workspace layout (bytes), total ~192 MB of bf16 intermediates.
  char* ws = (char*)d_ws;
  bf16* xb  = (bf16*)(ws);                    // 8192x2048       (32 MB)
  bf16* Wqb = (bf16*)(ws + 33554432);         // 2048x2048 W^T   ( 8 MB)
  bf16* Wkb = (bf16*)(ws + 41943040);
  bf16* Wvb = (bf16*)(ws + 50331648);
  bf16* Wob = (bf16*)(ws + 58720256);
  bf16* Qb  = (bf16*)(ws + 67108864);         // [b,h,s,hd]      (32 MB)
  bf16* Kb  = (bf16*)(ws + 100663296);        // [b,h,s,hd]
  bf16* Vt  = (bf16*)(ws + 134217728);        // [b,h,hd,s]
  bf16* Cb  = (bf16*)(ws + 167772160);        // ctx [b,s,D]

  cvt_f32_bf16<<<16384, 256, 0, stream>>>(x, xb, Bn * Sn * Dn);
  cvt_transpose<<<4096, 256, 0, stream>>>(Wq, Wqb);
  cvt_transpose<<<4096, 256, 0, stream>>>(Wk, Wkb);
  cvt_transpose<<<4096, 256, 0, stream>>>(Wv, Wvb);
  cvt_transpose<<<4096, 256, 0, stream>>>(Wo, Wob);

  dim3 ggrid((Mtot / 128) * (Dn / 128));      // 1024 blocks
  gemm_wmma<0><<<ggrid, 256, 0, stream>>>(xb, Wqb, (void*)Qb, nullptr);
  gemm_wmma<0><<<ggrid, 256, 0, stream>>>(xb, Wkb, (void*)Kb, nullptr);
  gemm_wmma<1><<<ggrid, 256, 0, stream>>>(xb, Wvb, (void*)Vt, nullptr);

  attn_fused<<<2048, 128, 0, stream>>>(Qb, Kb, Vt, Cb);

  gemm_wmma<2><<<ggrid, 256, 0, stream>>>(Cb, Wob, (void*)out, bo);
}
